// SwinTransformerBlock3D_87531433493083
// MI455X (gfx1250) — matrix operand order repaired and checked by reading, not compiled, not run on hardware
//
#include <hip/hip_runtime.h>

typedef _Float16 f16;
typedef __attribute__((ext_vector_type(16))) _Float16 v16h;
typedef __attribute__((ext_vector_type(8)))  float    v8f;

#define DIM     384
#define NHEADS  8
#define HD      48
#define HIDDEN  1536
#define NTOK    65536
#define SCALE_Q 0.14433756729740643f  // 48^-0.5
#define EPSV    1e-5f

// Load a 16-element f16 fragment as two 16B chunks (off2 = element offset of 2nd chunk)
static __device__ __forceinline__ v16h ld_frag(const f16* p, int off2) {
  union { uint4 u[2]; v16h v; } f;
  f.u[0] = *reinterpret_cast<const uint4*>(p);
  f.u[1] = *reinterpret_cast<const uint4*>(p + off2);
  return f.v;
}

// One per-lane async 16B copy global -> LDS (tracked by ASYNCcnt)
static __device__ __forceinline__ void async_cp16(f16* ldsdst, const f16* gsrc) {
  asm volatile("global_load_async_to_lds_b128 %0, %1, off"
               :: "v"((unsigned)(size_t)ldsdst), "v"(gsrc)
               : "memory");
}

// ---------------------------------------------------------------------------
// Weight convert + transpose: w[K][N] fp32 -> bt[N][K] f16 (k contiguous)
// ---------------------------------------------------------------------------
__global__ void wconv_t(const float* __restrict__ w, f16* __restrict__ bt, int K, int N) {
  int i = blockIdx.x * 256 + threadIdx.x;
  if (i < K * N) {
    int n = i / K, k = i % K;
    bt[i] = (f16)w[(size_t)k * N + n];
  }
}

// ---------------------------------------------------------------------------
// LN1 + window partition. Token order: win = ((b*8+dblk)*8+hblk)*8+wblk,
// tok = dd*16+hh*4+ww. One block handles 16 tokens (one dd-slice of a window).
// Writes f16 normalized activations and fp32 shortcut (x2 buffer).
// ---------------------------------------------------------------------------
__global__ __launch_bounds__(256) void ln1_partition(
    const float* __restrict__ x, const float* __restrict__ g, const float* __restrict__ bta,
    f16* __restrict__ xn, float* __restrict__ x2) {
  __shared__ float xs[16 * 385];
  const int tid = threadIdx.x;
  const int blk = blockIdx.x;          // 4096 blocks
  const int win = blk >> 2, dd = blk & 3;
  const int b = win >> 9, dblk = (win >> 6) & 7, hblk = (win >> 3) & 7, wblk = win & 7;

  // coalesced load: 384 channels x 4 hh rows x float4 (ww)
  #pragma unroll
  for (int i = 0; i < 6; i++) {
    int ch = tid + i * 256;            // 0..1535
    int c = ch >> 2, hh = ch & 3;
    const float4 v = *reinterpret_cast<const float4*>(
        &x[((size_t)(b * DIM + c)) * 32768 +
           (size_t)((dblk * 4 + dd) * 1024 + (hblk * 4 + hh) * 32 + wblk * 4)]);
    int tb = hh * 4;
    xs[(tb + 0) * 385 + c] = v.x;
    xs[(tb + 1) * 385 + c] = v.y;
    xs[(tb + 2) * 385 + c] = v.z;
    xs[(tb + 3) * 385 + c] = v.w;
  }
  __syncthreads();

  const int lane = tid & 31, wave = tid >> 5;
  #pragma unroll
  for (int tt = 0; tt < 2; tt++) {
    int tok16 = wave * 2 + tt;         // 0..15
    float s = 0.f, s2 = 0.f;
    #pragma unroll
    for (int j = 0; j < 12; j++) {
      float v = xs[tok16 * 385 + lane + j * 32];
      s += v; s2 += v * v;
    }
    #pragma unroll
    for (int o = 16; o >= 1; o >>= 1) { s += __shfl_xor(s, o, 32); s2 += __shfl_xor(s2, o, 32); }
    float mu = s * (1.f / 384.f);
    float rinv = rsqrtf(s2 * (1.f / 384.f) - mu * mu + EPSV);
    size_t base = ((size_t)win * 64 + dd * 16 + tok16) * DIM;
    #pragma unroll
    for (int j = 0; j < 12; j++) {
      int c = lane + j * 32;
      float v = xs[tok16 * 385 + c];
      x2[base + c] = v;
      xn[base + c] = (f16)((v - mu) * rinv * g[c] + bta[c]);
    }
  }
}

// ---------------------------------------------------------------------------
// LN2: one wave per token, x2 fp32 [NTOK][384] -> xn2 f16
// ---------------------------------------------------------------------------
__global__ __launch_bounds__(256) void ln2_kernel(
    const float* __restrict__ x2, const float* __restrict__ g, const float* __restrict__ bta,
    f16* __restrict__ xn2) {
  const int tok = blockIdx.x * 8 + (threadIdx.x >> 5);
  const int lane = threadIdx.x & 31;
  size_t base = (size_t)tok * DIM;
  float v[12], s = 0.f, s2 = 0.f;
  #pragma unroll
  for (int j = 0; j < 12; j++) { v[j] = x2[base + lane + j * 32]; s += v[j]; s2 += v[j] * v[j]; }
  #pragma unroll
  for (int o = 16; o >= 1; o >>= 1) { s += __shfl_xor(s, o, 32); s2 += __shfl_xor(s2, o, 32); }
  float mu = s * (1.f / 384.f);
  float rinv = rsqrtf(s2 * (1.f / 384.f) - mu * mu + EPSV);
  #pragma unroll
  for (int j = 0; j < 12; j++) {
    int c = lane + j * 32;
    xn2[base + c] = (f16)((v[j] - mu) * rinv * g[c] + bta[c]);
  }
}

// ---------------------------------------------------------------------------
// Tiled WMMA GEMM: C[M,N] = A[M,K](f16) * Bt[N,K](f16)^T + bias, fused epilogue.
// 256 threads / 8 waves; tile M=64, N=128; K step 32.
// Double-buffered LDS staging via GLOBAL_LOAD_ASYNC_TO_LDS_B128 (ASYNCcnt):
// issue next tile's copies, wait asynccnt<=3 (3 ops/wave outstanding = just the
// new ones; async loads retire in order), barrier, WMMA on current buffer.
// EP=0: qkv (scale q cols) -> f16 out   EP=1: proj, x2 += v (residual)
// EP=2: fc1 + exact GELU -> f16 out     EP=3: fc2 + residual + NCDHW scatter
// ---------------------------------------------------------------------------
template <int EP>
__global__ __launch_bounds__(256) void gemm_wmma(
    const f16* __restrict__ A, const f16* __restrict__ Bt,
    const float* __restrict__ bias, int K, int N,
    f16* __restrict__ outh, float* __restrict__ accbuf, float* __restrict__ dout) {
  __shared__ f16 As[2][64 * 40];    // padded stride 40 (80B) to avoid bank conflicts
  __shared__ f16 Bs[2][128 * 40];
  const int tid = threadIdx.x;
  const int lane = tid & 31, wave = tid >> 5;
  const int mw = wave & 3, nw = wave >> 2;
  const int blockN = blockIdx.x * 128;
  const int blockM = blockIdx.y * 64;
  const int lanelo = lane & 15, hi = lane >> 4;

  v8f acc[4] = {};
  const int arow = tid >> 2, aseg = tid & 3;   // A: 64 rows x 4 16B segs
  const int brow = tid >> 2, bseg = tid & 3;   // B: 2 x (64 rows x 4 16B segs)

  // issue one K-tile's async copies into buffer bf (3 instructions per wave)
  auto stage = [&](int bf, int kk) {
    async_cp16(&As[bf][arow * 40 + aseg * 8],
               &A[(size_t)(blockM + arow) * K + kk + aseg * 8]);
    async_cp16(&Bs[bf][brow * 40 + bseg * 8],
               &Bt[(size_t)(blockN + brow) * K + kk + bseg * 8]);
    async_cp16(&Bs[bf][(brow + 64) * 40 + bseg * 8],
               &Bt[(size_t)(blockN + 64 + brow) * K + kk + bseg * 8]);
  };

  stage(0, 0);
  int buf = 0;
  for (int kk = 0; kk < K; kk += 32) {
    if (kk + 32 < K) {
      stage(buf ^ 1, kk + 32);
      asm volatile("s_wait_asynccnt 0x3" ::: "memory");  // current tile landed
    } else {
      asm volatile("s_wait_asynccnt 0x0" ::: "memory");
    }
    __syncthreads();
    // A frag: lanes 0-15 K=[0..7,16..23], lanes 16-31 K=[8..15,24..31]
    v16h af = ld_frag(&As[buf][(mw * 16 + lanelo) * 40 + hi * 8], 16);
    #pragma unroll
    for (int t = 0; t < 4; t++) {
      // B frag: lanes 0-15 K=[0..7,16..23], lanes 16-31 K=[8..15,24..31]
      v16h bf16 = ld_frag(&Bs[buf][(nw * 64 + t * 16 + lanelo) * 40 + hi * 8], 16);
      acc[t] = __builtin_amdgcn_wmma_f32_16x16x32_f16(false, af, false, bf16, (short)0,
                                                      acc[t], false, false);
    }
    __syncthreads();   // everyone done reading before this buffer is re-staged
    buf ^= 1;
  }

  #pragma unroll
  for (int t = 0; t < 4; t++) {
    int n = blockN + nw * 64 + t * 16 + lanelo;
    float bv = bias[n];
    #pragma unroll
    for (int r = 0; r < 8; r++) {
      int m = blockM + mw * 16 + hi * 8 + r;
      float v = acc[t][r] + bv;
      if (EP == 0) {
        if (n < DIM) v *= SCALE_Q;                      // pre-scale q
        outh[(size_t)m * N + n] = (f16)v;
      } else if (EP == 1) {
        accbuf[(size_t)m * DIM + n] += v;               // x2 = shortcut + proj(attn)
      } else if (EP == 2) {
        float gv = 0.5f * v * (1.0f + erff(v * 0.70710678118f));  // exact GELU
        outh[(size_t)m * N + n] = (f16)gv;
      } else {
        float xv = accbuf[(size_t)m * DIM + n] + v;     // final residual
        int win = m >> 6, tok = m & 63;
        int b = win >> 9, dblk = (win >> 6) & 7, hblk = (win >> 3) & 7, wblk = win & 7;
        int dd = tok >> 4, hh = (tok >> 2) & 3, ww = tok & 3;
        size_t off = ((size_t)(b * DIM + n)) * 32768 +
                     (size_t)((dblk * 4 + dd) * 1024 + (hblk * 4 + hh) * 32 + wblk * 4 + ww);
        dout[off] = xv;
      }
    }
  }
}

// ---------------------------------------------------------------------------
// Windowed attention: one block per (window, head). 4 waves.
// S = Q K^T (WMMA, hd 48 zero-padded to 64), + rel-pos bias, fp32 softmax,
// O = P V (WMMA, V staged transposed).
// ---------------------------------------------------------------------------
__global__ __launch_bounds__(128) void attn_win(
    const f16* __restrict__ qkv, const float* __restrict__ btab, f16* __restrict__ aout) {
  __shared__ f16 qs[64 * 72];     // Q rows (later reused for P)
  __shared__ f16 ks[64 * 72];     // K rows
  __shared__ f16 vt[64 * 72];     // V transposed: vt[d][tok]
  __shared__ float sb[64 * 68];   // fp32 scores
  const int win = blockIdx.x, head = blockIdx.y;
  const int tid = threadIdx.x, lane = tid & 31, wave = tid >> 5;
  const int lanelo = lane & 15, hi = lane >> 4;

  // zero the padded K-range cols 48..63 of q/k (hd=48 -> K=64)
  for (int i = tid; i < 64 * 16; i += 128) {
    int r = i >> 4, c = 48 + (i & 15);
    qs[r * 72 + c] = (f16)0.f;
    ks[r * 72 + c] = (f16)0.f;
  }
  // stage q,k (row-major) and v (transposed)
  size_t wbase = (size_t)win * 64 * 1152 + head * HD;
  for (int ch = tid; ch < 768; ch += 128) {
    int tok = ch / 12, seg = ch % 12;
    size_t gbase = wbase + (size_t)tok * 1152 + seg * 4;
    *reinterpret_cast<uint2*>(&qs[tok * 72 + seg * 4]) =
        *reinterpret_cast<const uint2*>(&qkv[gbase]);
    *reinterpret_cast<uint2*>(&ks[tok * 72 + seg * 4]) =
        *reinterpret_cast<const uint2*>(&qkv[gbase + 384]);
    const f16* vp = &qkv[gbase + 768];
    #pragma unroll
    for (int j = 0; j < 4; j++) vt[(seg * 4 + j) * 72 + tok] = vp[j];
  }
  __syncthreads();

  // S = q @ k^T  (B frag reads k rows directly: B[kdim][n] = k[n][kdim])
  v8f accS[4] = {};
  #pragma unroll
  for (int kk = 0; kk < 64; kk += 32) {
    v16h af = ld_frag(&qs[(wave * 16 + lanelo) * 72 + kk + hi * 8], 16);
    #pragma unroll
    for (int t = 0; t < 4; t++) {
      v16h bf = ld_frag(&ks[(t * 16 + lanelo) * 72 + kk + hi * 8], 16);
      accS[t] = __builtin_amdgcn_wmma_f32_16x16x32_f16(false, af, false, bf, (short)0,
                                                       accS[t], false, false);
    }
  }
  #pragma unroll
  for (int t = 0; t < 4; t++)
    #pragma unroll
    for (int r = 0; r < 8; r++)
      sb[(wave * 16 + hi * 8 + r) * 68 + t * 16 + lanelo] = accS[t][r];
  __syncthreads();

  // softmax with rel-pos bias (REL_IDX recomputed: 4x4x4 geometry, table 7x7x7)
  if (tid < 64) {
    int n = tid;
    int dn = n >> 4, hn = (n >> 2) & 3, wn = n & 3;
    float mx = -1e30f;
    for (int m = 0; m < 64; m++) {
      int dm = m >> 4, hm = (m >> 2) & 3, wm = m & 3;
      int idx = (dn - dm + 3) * 49 + (hn - hm + 3) * 7 + (wn - wm + 3);
      float v = sb[n * 68 + m] + btab[idx * NHEADS + head];
      sb[n * 68 + m] = v;
      mx = fmaxf(mx, v);
    }
    float sum = 0.f;
    for (int m = 0; m < 64; m++) {
      float e = __expf(sb[n * 68 + m] - mx);
      sb[n * 68 + m] = e;
      sum += e;
    }
    float rinv = 1.f / sum;
    for (int m = 0; m < 64; m++) qs[n * 72 + m] = (f16)(sb[n * 68 + m] * rinv);  // P into qs
  }
  __syncthreads();

  // O = P @ V  (B frag from vt rows; only 3 N-subtiles since hd=48)
  v8f accO[3] = {};
  #pragma unroll
  for (int kk = 0; kk < 64; kk += 32) {
    v16h af = ld_frag(&qs[(wave * 16 + lanelo) * 72 + kk + hi * 8], 16);
    #pragma unroll
    for (int t = 0; t < 3; t++) {
      v16h bf = ld_frag(&vt[(t * 16 + lanelo) * 72 + kk + hi * 8], 16);
      accO[t] = __builtin_amdgcn_wmma_f32_16x16x32_f16(false, af, false, bf, (short)0,
                                                       accO[t], false, false);
    }
  }
  #pragma unroll
  for (int t = 0; t < 3; t++)
    #pragma unroll
    for (int r = 0; r < 8; r++) {
      int m = wave * 16 + hi * 8 + r;
      int d = t * 16 + lanelo;
      aout[((size_t)win * 64 + m) * DIM + head * HD + d] = (f16)accO[t][r];
    }
}

// ---------------------------------------------------------------------------
extern "C" void kernel_launch(void* const* d_in, const int* in_sizes, int n_in,
                              void* d_out, int out_size, void* d_ws, size_t ws_size,
                              hipStream_t stream) {
  const float* x       = (const float*)d_in[0];
  const float* n1g     = (const float*)d_in[1];
  const float* n1b     = (const float*)d_in[2];
  const float* qkv_w   = (const float*)d_in[3];
  const float* qkv_b   = (const float*)d_in[4];
  const float* rel_tab = (const float*)d_in[5];
  const float* proj_w  = (const float*)d_in[6];
  const float* proj_b  = (const float*)d_in[7];
  const float* n2g     = (const float*)d_in[8];
  const float* n2b     = (const float*)d_in[9];
  const float* fc1_w   = (const float*)d_in[10];
  const float* fc1_b   = (const float*)d_in[11];
  const float* fc2_w   = (const float*)d_in[12];
  const float* fc2_b   = (const float*)d_in[13];
  float* out = (float*)d_out;

  char* ws = (char*)d_ws;
  size_t off = 0;
  auto alloc = [&](size_t bytes) {
    void* p = ws + off;
    off = (off + bytes + 255) & ~(size_t)255;
    return p;
  };
  f16*   wt_qkv = (f16*)alloc((size_t)1152 * 384 * 2);
  f16*   wt_prj = (f16*)alloc((size_t)384 * 384 * 2);
  f16*   wt_fc1 = (f16*)alloc((size_t)1536 * 384 * 2);
  f16*   wt_fc2 = (f16*)alloc((size_t)384 * 1536 * 2);
  f16*   xn     = (f16*)alloc((size_t)NTOK * 384 * 2);
  float* x2     = (float*)alloc((size_t)NTOK * 384 * 4);
  f16*   qkvb   = (f16*)alloc((size_t)NTOK * 1152 * 2);
  f16*   aoutb  = (f16*)alloc((size_t)NTOK * 384 * 2);
  f16*   xn2    = (f16*)alloc((size_t)NTOK * 384 * 2);
  f16*   hbuf   = (f16*)alloc((size_t)NTOK * 1536 * 2);

  // weights -> f16, transposed to [N][K]
  wconv_t<<<(1152 * 384 + 255) / 256, 256, 0, stream>>>(qkv_w, wt_qkv, 384, 1152);
  wconv_t<<<(384 * 384 + 255) / 256, 256, 0, stream>>>(proj_w, wt_prj, 384, 384);
  wconv_t<<<(1536 * 384 + 255) / 256, 256, 0, stream>>>(fc1_w, wt_fc1, 384, 1536);
  wconv_t<<<(384 * 1536 + 255) / 256, 256, 0, stream>>>(fc2_w, wt_fc2, 1536, 384);

  // LN1 + partition (shortcut into x2)
  ln1_partition<<<4096, 256, 0, stream>>>(x, n1g, n1b, xn, x2);

  // QKV projection
  gemm_wmma<0><<<dim3(1152 / 128, NTOK / 64), 256, 0, stream>>>(
      xn, wt_qkv, qkv_b, 384, 1152, qkvb, nullptr, nullptr);

  // windowed attention
  attn_win<<<dim3(1024, NHEADS), 128, 0, stream>>>(qkvb, rel_tab, aoutb);

  // proj + residual (x2 += proj)
  gemm_wmma<1><<<dim3(384 / 128, NTOK / 64), 256, 0, stream>>>(
      aoutb, wt_prj, proj_b, 384, 384, nullptr, x2, nullptr);

  // LN2
  ln2_kernel<<<NTOK / 8, 256, 0, stream>>>(x2, n2g, n2b, xn2);

  // fc1 + GELU
  gemm_wmma<2><<<dim3(1536 / 128, NTOK / 64), 256, 0, stream>>>(
      xn2, wt_fc1, fc1_b, 384, 1536, hbuf, nullptr, nullptr);

  // fc2 + residual + scatter to (B,C,D,H,W)
  gemm_wmma<3><<<dim3(384 / 128, NTOK / 64), 256, 0, stream>>>(
      hbuf, wt_fc2, fc2_b, 1536, 384, nullptr, x2, out);
}
